// Transducer_14817637171800
// MI455X (gfx1250) — matrix operand order repaired and checked by reading, not compiled, not run on hardware
//
#include <hip/hip_runtime.h>
#include <hip/hip_bf16.h>

typedef __attribute__((ext_vector_type(16))) _Float16 v16h;
typedef __attribute__((ext_vector_type(8)))  float    v8f;

constexpr int Bc = 2, Tc = 256, Uc = 64, Dc = 512;
constexpr int INNERc = 512, VOCABc = 4096;
constexpr int BLK_N = 256;  // 8 waves x 2 slices x 16 vocab cols
constexpr int BLK_M = 64;   // = Uc, one (b,t) per block

// ---------------------------------------------------------------------------
// Kernel 1: enc_proj (B*T x INNER) and dec_proj (B*U x INNER, with b1 folded)
// ---------------------------------------------------------------------------
__global__ void proj_kernel(const float* __restrict__ enc,
                            const float* __restrict__ dec,
                            const float* __restrict__ W1,
                            const float* __restrict__ b1,
                            float* __restrict__ ep,
                            float* __restrict__ dp) {
    int idx = blockIdx.x * blockDim.x + threadIdx.x;  // one per output elem
    int i   = idx & (INNERc - 1);
    int row = idx >> 9;  // log2(INNER)
    const int NE = Bc * Tc;

    const float4* x;
    const float4* w;
    float acc;
    float* outp;
    if (row < NE) {
        x    = (const float4*)(enc + (size_t)row * Dc);
        w    = (const float4*)(W1 + (size_t)i * (2 * Dc));
        acc  = 0.0f;
        outp = ep + (size_t)row * INNERc + i;
    } else {
        int r = row - NE;
        x    = (const float4*)(dec + (size_t)r * Dc);
        w    = (const float4*)(W1 + (size_t)i * (2 * Dc) + Dc);
        acc  = b1[i];  // fold bias into dec_proj
        outp = dp + (size_t)r * INNERc + i;
    }
#pragma unroll 4
    for (int k = 0; k < Dc / 4; ++k) {
        float4 a = x[k], b = w[k];
        acc += a.x * b.x + a.y * b.y + a.z * b.z + a.w * b.w;
    }
    *outp = acc;
}

// ---------------------------------------------------------------------------
// Kernel 2: W2 f32 -> f16 (done once; 4 MB stays L2-resident)
// ---------------------------------------------------------------------------
__global__ void w2cvt_kernel(const float* __restrict__ W2,
                             _Float16* __restrict__ W2h) {
    int idx = blockIdx.x * blockDim.x + threadIdx.x;  // one per 4 elems
    float4 v = *(const float4*)(W2 + (size_t)idx * 4);
    union { _Float16 h[4]; uint2 u; } p;
    p.h[0] = (_Float16)v.x; p.h[1] = (_Float16)v.y;
    p.h[2] = (_Float16)v.z; p.h[3] = (_Float16)v.w;
    *(uint2*)(W2h + (size_t)idx * 4) = p.u;
}

// ---------------------------------------------------------------------------
// Kernel 3: h = tanh(ep[bt] + dp[bu]) in f16, materialized once (33.5 MB)
// ---------------------------------------------------------------------------
__global__ void h_kernel(const float* __restrict__ ep,
                         const float* __restrict__ dp,
                         _Float16* __restrict__ hb) {
    int idx = blockIdx.x * blockDim.x + threadIdx.x;  // one per 4 elems
    int k4 = idx & (INNERc / 4 - 1);   // 0..127
    int m  = idx >> 7;                 // flat (b,t,u) row
    int u  = m & (Uc - 1);
    int bt = m >> 6;                   // log2(U)
    int b  = bt >> 8;                  // log2(T)

    float4 e = *(const float4*)(ep + (size_t)bt * INNERc + k4 * 4);
    float4 d = *(const float4*)(dp + (size_t)(b * Uc + u) * INNERc + k4 * 4);
    union { _Float16 h[4]; uint2 u2; } p;
    p.h[0] = (_Float16)tanhf(e.x + d.x);
    p.h[1] = (_Float16)tanhf(e.y + d.y);
    p.h[2] = (_Float16)tanhf(e.z + d.z);
    p.h[3] = (_Float16)tanhf(e.w + d.w);
    *(uint2*)(hb + (size_t)idx * 4) = p.u2;
}

// ---------------------------------------------------------------------------
// Kernel 4: logits = h @ W2h^T + b2 via v_wmma_f32_16x16x32_f16
// Block: one (b,t) -> M=64 rows (all u), N=256 vocab cols, K=512.
// Each wave owns two 16-wide N slices (2 B-frags, 8 accumulators) so the
// 64KB LDS h-tile and each A-fragment read feed 2x the WMMAs.
// ---------------------------------------------------------------------------
__device__ __forceinline__ int swz(int r, int c8) {
    // 8-half (16B) chunk granularity; xor rotate chunks by row
    return r * INNERc + ((c8 ^ (r & 15)) << 3);
}

__global__ __launch_bounds__(256) void joint_gemm(
        const _Float16* __restrict__ hbuf,
        const _Float16* __restrict__ W2h,
        const float* __restrict__ b2,
        float* __restrict__ out) {
    __shared__ _Float16 hs[BLK_M * INNERc];  // 64 x 512 f16 = 65536 B

    const int bt    = blockIdx.y;            // 0 .. B*T-1
    const int n0    = blockIdx.x * BLK_N;    // vocab tile start
    const size_t m0 = (size_t)bt * Uc;       // flat row base in hbuf

    // ---- stage h tile global -> LDS (contiguous 64x512 halfs) ----
    {
        const uint4* src = (const uint4*)(hbuf + m0 * INNERc);
        int tid = threadIdx.x;
#pragma unroll
        for (int it = 0; it < 16; ++it) {
            int li = tid + it * 256;     // 0..4095 chunk index
            int r  = li >> 6;            // row 0..63
            int c8 = li & 63;            // 16B chunk within row
            uint4 v = src[(size_t)r * 64 + c8];
            *(uint4*)(&hs[swz(r, c8)]) = v;
        }
    }
    __syncthreads();

    const int wave = threadIdx.x >> 5;   // 0..7
    const int lane = threadIdx.x & 31;
    const int ncol = lane & 15;          // N within 16-wide slice
    const int kh   = lane >> 4;          // 0/1: K-half select
    const int kh8  = kh * 8;
    const int nsubA = n0 + wave * 32;    // first 16-wide slice
    const int nsubB = nsubA + 16;        // second 16-wide slice

    // B fragment sources: lane owns W2 row (nsub+ncol), K-contiguous
    const _Float16* w2pA = W2h + (size_t)(nsubA + ncol) * INNERc + kh * 16;
    const _Float16* w2pB = W2h + (size_t)(nsubB + ncol) * INNERc + kh * 16;

    v8f accA0 = {}, accA1 = {}, accA2 = {}, accA3 = {};
    v8f accB0 = {}, accB1 = {}, accB2 = {}, accB3 = {};

    for (int kk = 0; kk < INNERc; kk += 32) {
        v16h bfA = *(const v16h*)(w2pA + kk);
        v16h bfB = *(const v16h*)(w2pB + kk);

        union AU { uint4 q[2]; v16h v; };
        AU a0, a1, a2, a3;
        const int j0 = (kk + kh * 16) >> 3;       // first chunk
        const int j1 = (kk + kh * 16 + 8) >> 3;  // second chunk
        const int rl = lane & 15;

        a0.q[0] = *(const uint4*)&hs[swz(rl,      j0)];
        a0.q[1] = *(const uint4*)&hs[swz(rl,      j1)];
        a1.q[0] = *(const uint4*)&hs[swz(rl + 16, j0)];
        a1.q[1] = *(const uint4*)&hs[swz(rl + 16, j1)];
        a2.q[0] = *(const uint4*)&hs[swz(rl + 32, j0)];
        a2.q[1] = *(const uint4*)&hs[swz(rl + 32, j1)];
        a3.q[0] = *(const uint4*)&hs[swz(rl + 48, j0)];
        a3.q[1] = *(const uint4*)&hs[swz(rl + 48, j1)];

        accA0 = __builtin_amdgcn_wmma_f32_16x16x32_f16(false, a0.v, false, bfA,
                                                       (short)0, accA0, false, false);
        accB0 = __builtin_amdgcn_wmma_f32_16x16x32_f16(false, a0.v, false, bfB,
                                                       (short)0, accB0, false, false);
        accA1 = __builtin_amdgcn_wmma_f32_16x16x32_f16(false, a1.v, false, bfA,
                                                       (short)0, accA1, false, false);
        accB1 = __builtin_amdgcn_wmma_f32_16x16x32_f16(false, a1.v, false, bfB,
                                                       (short)0, accB1, false, false);
        accA2 = __builtin_amdgcn_wmma_f32_16x16x32_f16(false, a2.v, false, bfA,
                                                       (short)0, accA2, false, false);
        accB2 = __builtin_amdgcn_wmma_f32_16x16x32_f16(false, a2.v, false, bfB,
                                                       (short)0, accB2, false, false);
        accA3 = __builtin_amdgcn_wmma_f32_16x16x32_f16(false, a3.v, false, bfA,
                                                       (short)0, accA3, false, false);
        accB3 = __builtin_amdgcn_wmma_f32_16x16x32_f16(false, a3.v, false, bfB,
                                                       (short)0, accB3, false, false);
    }

    // ---- epilogue: C layout VGPR r -> row (r + kh*8), col = lane&15 ----
    const float bvA = b2[nsubA + ncol];
    const float bvB = b2[nsubB + ncol];
    const size_t obA = m0 * VOCABc + (size_t)(nsubA + ncol);
    const size_t obB = m0 * VOCABc + (size_t)(nsubB + ncol);
#pragma unroll
    for (int r = 0; r < 8; ++r) {
        const size_t r0 = (size_t)( 0 + r + kh8) * VOCABc;
        const size_t r1 = (size_t)(16 + r + kh8) * VOCABc;
        const size_t r2 = (size_t)(32 + r + kh8) * VOCABc;
        const size_t r3 = (size_t)(48 + r + kh8) * VOCABc;
        out[obA + r0] = accA0[r] + bvA;
        out[obB + r0] = accB0[r] + bvB;
        out[obA + r1] = accA1[r] + bvA;
        out[obB + r1] = accB1[r] + bvB;
        out[obA + r2] = accA2[r] + bvA;
        out[obB + r2] = accB2[r] + bvB;
        out[obA + r3] = accA3[r] + bvA;
        out[obB + r3] = accB3[r] + bvB;
    }
}

// ---------------------------------------------------------------------------
extern "C" void kernel_launch(void* const* d_in, const int* in_sizes, int n_in,
                              void* d_out, int out_size, void* d_ws, size_t ws_size,
                              hipStream_t stream) {
    const float* enc = (const float*)d_in[0];  // (B,T,D)
    const float* dec = (const float*)d_in[1];  // (B,U,D)
    const float* W1  = (const float*)d_in[2];  // (INNER, 2D)
    const float* b1  = (const float*)d_in[3];  // (INNER)
    const float* W2  = (const float*)d_in[4];  // (VOCAB, INNER)
    const float* b2  = (const float*)d_in[5];  // (VOCAB)
    float* out = (float*)d_out;                // (B,T,U,VOCAB)

    // workspace layout (all 16B aligned)
    float*    ep  = (float*)d_ws;                          // 512*512 f32
    float*    dp  = ep + (size_t)Bc * Tc * INNERc;         // 128*512 f32
    _Float16* w2h = (_Float16*)(dp + (size_t)Bc * Uc * INNERc);
    _Float16* hb  = w2h + (size_t)VOCABc * INNERc;         // 32768*512 f16

    // 1) projections (+ b1 folded into dec side)
    {
        int total = (Bc * Tc + Bc * Uc) * INNERc;  // 327680
        proj_kernel<<<total / 256, 256, 0, stream>>>(enc, dec, W1, b1, ep, dp);
    }
    // 2) W2 -> f16
    {
        int total = VOCABc * INNERc / 4;  // 524288
        w2cvt_kernel<<<total / 256, 256, 0, stream>>>(W2, w2h);
    }
    // 3) h = tanh(ep + dp) in f16, once
    {
        int total = Bc * Tc * Uc * INNERc / 4;  // 4194304
        h_kernel<<<total / 256, 256, 0, stream>>>(ep, dp, hb);
    }
    // 4) big GEMM with WMMA
    {
        dim3 grid(VOCABc / BLK_N, Bc * Tc);  // 16 x 512
        joint_gemm<<<grid, 256, 0, stream>>>(hb, w2h, b2, out);
    }
}
